// ConvNeXtParallelMoELoRA_40407052321162
// MI455X (gfx1250) — compile-verified, matrix operand order checked by reading, not run on hardware
//
#include <hip/hip_runtime.h>

// MI455X / gfx1250 fused ConvNeXt-MLP + parallel LoRA-MoE.
// bf16 WMMA (16x16x32) with hi/lo split-precision (3-term) for ~fp32 accuracy.

typedef __attribute__((ext_vector_type(16))) __bf16 v16bf;
typedef __attribute__((ext_vector_type(8)))  float  v8f;

#define D_IN   512
#define D_HID  2048
#define T_TOK  50176
#define M_TILE 64
#define CHUNK  64
#define XLD    520   // 512 + 8 ushort pad -> 1040B rows (16B aligned, bank-conflict free)
#define HLD    72    // 64 + 8 pad -> 144B rows
#define GLD    26

__device__ __forceinline__ unsigned short f2bf(float f) {
  unsigned u = __float_as_uint(f);
  return (unsigned short)((u + 0x7fffu + ((u >> 16) & 1u)) >> 16); // RNE
}
__device__ __forceinline__ float bf2f(unsigned short b) {
  return __uint_as_float(((unsigned)b) << 16);
}
__device__ __forceinline__ float gelu_exact(float v) {
  return 0.5f * v * (1.0f + erff(v * 0.70710678118654752f));
}

union BFrag { v16bf v; unsigned short u[16]; uint4 q[2]; };

// Pre-swizzle a row-major KxN f32 weight matrix into per-(16x32)-tile,
// per-lane fragment order as bf16 hi / lo buffers. Tile t = (n0/16)*(K/32)+(k0/32),
// 512 ushorts per tile, lane-contiguous 16-element groups -> b128 loads in hot loop.
__global__ void swizzle_w_kernel(const float* __restrict__ W, int K, int N,
                                 unsigned short* __restrict__ oh,
                                 unsigned short* __restrict__ ol) {
  int idx = blockIdx.x * 256 + threadIdx.x;
  if (idx >= K * N) return;
  int tile = idx >> 9, rem = idx & 511;
  int ln = rem >> 4, j = rem & 15;
  int ktiles = K >> 5;
  int k0 = (tile % ktiles) << 5;
  int n0 = (tile / ktiles) << 4;
  int k = k0 + ((ln >> 4) << 3) + (j & 7) + ((j >> 3) << 4);
  int n = n0 + (ln & 15);
  float v = W[(size_t)k * N + n];
  unsigned short hi = f2bf(v);
  oh[idx] = hi;
  ol[idx] = f2bf(v - bf2f(hi));
}

__global__ __launch_bounds__(256) void fused_mlp_moe_kernel(
    const float* __restrict__ x,
    const float* __restrict__ b1,
    const float* __restrict__ b2,
    const float* __restrict__ w_down,   // [3][512][8]
    const float* __restrict__ w_up,     // [3][8][512] == [24][512]
    const float* __restrict__ tkp,      // [T][2] f32
    const long long* __restrict__ tki,  // [T][2] i64
    const unsigned short* __restrict__ w1h,
    const unsigned short* __restrict__ w1l,
    const unsigned short* __restrict__ w2h,
    const unsigned short* __restrict__ w2l,
    float* __restrict__ out)
{
  __shared__ __align__(16) unsigned short XH[M_TILE][XLD];
  __shared__ __align__(16) unsigned short XL[M_TILE][XLD];
  __shared__ __align__(16) unsigned short HH[M_TILE][HLD];
  __shared__ __align__(16) unsigned short HL[M_TILE][HLD];
  __shared__ float GDW[M_TILE][GLD];

  const int tid   = threadIdx.x;
  const int lane  = tid & 31;
  const int wv    = tid >> 5;
  const int lrow  = lane & 15;
  const int lhalf = lane >> 4;
  const int t0    = blockIdx.x * M_TILE;

  // ---- stage x tile into LDS as bf16 hi/lo ----
  for (int i = tid; i < M_TILE * (D_IN / 4); i += 256) {
    int m  = i / (D_IN / 4);
    int c4 = i % (D_IN / 4);
    float4 v = ((const float4*)(x + (size_t)(t0 + m) * D_IN))[c4];
    int c = c4 * 4;
    float f[4] = {v.x, v.y, v.z, v.w};
#pragma unroll
    for (int j = 0; j < 4; ++j) {
      unsigned short hi = f2bf(f[j]);
      XH[m][c + j] = hi;
      XL[m][c + j] = f2bf(f[j] - bf2f(hi));
    }
  }
  __syncthreads();

  // ---- LoRA down-GEMM + gate folding: GDW[m][e*8+r] = gate_w * gelu(x . w_down) ----
  for (int o = tid; o < M_TILE * 24; o += 256) {
    int m = o / 24, er = o % 24;
    int e = er >> 3, rr = er & 7;
    const float* wd = w_down + (size_t)e * D_IN * 8 + rr;
    float s = 0.f;
    for (int d = 0; d < D_IN; ++d)
      s += (bf2f(XH[m][d]) + bf2f(XL[m][d])) * wd[(size_t)d * 8];
    size_t gb = (size_t)(t0 + m) * 2;
    float wgt = 0.f;
    if ((int)tki[gb]     == e) wgt += tkp[gb];
    if ((int)tki[gb + 1] == e) wgt += tkp[gb + 1];
    GDW[m][er] = wgt * gelu_exact(s);   // SCALING == 1.0
  }

  v8f acc[16] = {};                     // per-wave 16 rows x 256 cols f32

  const int mi1 = wv >> 1;              // stage-1 row band (0..3)
  const int nb1 = (wv & 1) * 2;         // stage-1 first n-tile in chunk
  const int mi2 = wv & 3;               // stage-2 row band
  const int nh  = wv >> 2;              // stage-2 n-half (0/1)
  const int am1 = mi1 * 16 + lrow;
  const int am2 = mi2 * 16 + lrow;

  for (int j0 = 0; j0 < D_HID; j0 += CHUNK) {
    __syncthreads();                    // HH/HL reuse hazard (and first-iter GDW/X fence)

    // ---- stage 1: Hc = gelu(x @ W1[:, j0:j0+64] + b1) ----
    v8f hacc[2] = {};
    for (int k0 = 0; k0 < D_IN; k0 += 32) {
      BFrag ah, al;
      const uint4* ph = (const uint4*)&XH[am1][k0 + lhalf * 8];
      ah.q[0] = ph[0]; ah.q[1] = ph[2];
      const uint4* pl = (const uint4*)&XL[am1][k0 + lhalf * 8];
      al.q[0] = pl[0]; al.q[1] = pl[2];
#pragma unroll
      for (int tt = 0; tt < 2; ++tt) {
        int n0 = j0 + (nb1 + tt) * 16;
        size_t tb = (((size_t)(n0 >> 4) * (D_IN / 32) + (k0 >> 5)) << 9) + (lane << 4);
        BFrag bh, bl;
        const uint4* pbh = (const uint4*)(w1h + tb);
        bh.q[0] = pbh[0]; bh.q[1] = pbh[1];
        const uint4* pbl = (const uint4*)(w1l + tb);
        bl.q[0] = pbl[0]; bl.q[1] = pbl[1];
        hacc[tt] = __builtin_amdgcn_wmma_f32_16x16x32_bf16(false, ah.v, false, bh.v, (short)0, hacc[tt], false, false);
        hacc[tt] = __builtin_amdgcn_wmma_f32_16x16x32_bf16(false, ah.v, false, bl.v, (short)0, hacc[tt], false, false);
        hacc[tt] = __builtin_amdgcn_wmma_f32_16x16x32_bf16(false, al.v, false, bh.v, (short)0, hacc[tt], false, false);
      }
    }
    // bias + exact GELU + hi/lo split-store into Hc
#pragma unroll
    for (int tt = 0; tt < 2; ++tt) {
      int ccol = (nb1 + tt) * 16 + lrow;
      float bv = b1[j0 + ccol];
#pragma unroll
      for (int q = 0; q < 8; ++q) {
        int rm = mi1 * 16 + q + 8 * lhalf;   // C/D layout: M = q + 8*(lane/16), N = lane%16
        float hval = gelu_exact(hacc[tt][q] + bv);
        unsigned short hi = f2bf(hval);
        HH[rm][ccol] = hi;
        HL[rm][ccol] = f2bf(hval - bf2f(hi));
      }
    }
    __syncthreads();

    // ---- stage 2: acc += Hc @ W2[j0:j0+64, :] ----
#pragma unroll
    for (int ks = 0; ks < 2; ++ks) {
      BFrag ah, al;
      const uint4* ph = (const uint4*)&HH[am2][ks * 32 + lhalf * 8];
      ah.q[0] = ph[0]; ah.q[1] = ph[2];
      const uint4* pl = (const uint4*)&HL[am2][ks * 32 + lhalf * 8];
      al.q[0] = pl[0]; al.q[1] = pl[2];
      int krow = j0 + ks * 32;
#pragma unroll
      for (int nt = 0; nt < 16; ++nt) {
        int n0 = nh * 256 + nt * 16;
        size_t tb = (((size_t)(n0 >> 4) * (D_HID / 32) + (krow >> 5)) << 9) + (lane << 4);
        BFrag bh, bl;
        const uint4* pbh = (const uint4*)(w2h + tb);
        bh.q[0] = pbh[0]; bh.q[1] = pbh[1];
        const uint4* pbl = (const uint4*)(w2l + tb);
        bl.q[0] = pbl[0]; bl.q[1] = pbl[1];
        acc[nt] = __builtin_amdgcn_wmma_f32_16x16x32_bf16(false, ah.v, false, bh.v, (short)0, acc[nt], false, false);
        acc[nt] = __builtin_amdgcn_wmma_f32_16x16x32_bf16(false, ah.v, false, bl.v, (short)0, acc[nt], false, false);
        acc[nt] = __builtin_amdgcn_wmma_f32_16x16x32_bf16(false, al.v, false, bh.v, (short)0, acc[nt], false, false);
      }
    }
  }

  // ---- LoRA up as one padded-K WMMA pass: acc += GDW(64x24,pad32) @ w_up(24x512) ----
  {
    BFrag ah, al;
#pragma unroll
    for (int j = 0; j < 16; ++j) {
      int k = lhalf * 8 + (j & 7) + ((j >> 3) << 4);
      float v = (k < 24) ? GDW[am2][k] : 0.f;
      unsigned short hi = f2bf(v);
      ah.u[j] = hi;
      al.u[j] = f2bf(v - bf2f(hi));
    }
#pragma unroll
    for (int nt = 0; nt < 16; ++nt) {
      int n = nh * 256 + nt * 16 + lrow;
      BFrag bh, bl;
#pragma unroll
      for (int j = 0; j < 16; ++j) {
        int k = lhalf * 8 + (j & 7) + ((j >> 3) << 4);
        float v = (k < 24) ? w_up[(size_t)k * D_IN + n] : 0.f;
        unsigned short hi = f2bf(v);
        bh.u[j] = hi;
        bl.u[j] = f2bf(v - bf2f(hi));
      }
      acc[nt] = __builtin_amdgcn_wmma_f32_16x16x32_bf16(false, ah.v, false, bh.v, (short)0, acc[nt], false, false);
      acc[nt] = __builtin_amdgcn_wmma_f32_16x16x32_bf16(false, ah.v, false, bl.v, (short)0, acc[nt], false, false);
      acc[nt] = __builtin_amdgcn_wmma_f32_16x16x32_bf16(false, al.v, false, bh.v, (short)0, acc[nt], false, false);
    }
  }

  // ---- epilogue: + b2, store fp32 ----
#pragma unroll
  for (int nt = 0; nt < 16; ++nt) {
    int n = nh * 256 + nt * 16 + lrow;
    float bv = b2[n];
#pragma unroll
    for (int q = 0; q < 8; ++q) {
      int row = t0 + mi2 * 16 + q + 8 * lhalf;
      out[(size_t)row * D_IN + n] = acc[nt][q] + bv;
    }
  }
}

extern "C" void kernel_launch(void* const* d_in, const int* in_sizes, int n_in,
                              void* d_out, int out_size, void* d_ws, size_t ws_size,
                              hipStream_t stream) {
  (void)in_sizes; (void)n_in; (void)out_size; (void)ws_size;
  const float* x      = (const float*)d_in[0];
  const float* W1     = (const float*)d_in[1];
  const float* b1     = (const float*)d_in[2];
  const float* W2     = (const float*)d_in[3];
  const float* b2     = (const float*)d_in[4];
  const float* w_down = (const float*)d_in[5];
  const float* w_up   = (const float*)d_in[6];
  /* d_in[7] = gate_probs (unused by reference output) */
  const float* tkp    = (const float*)d_in[8];
  const long long* tki = (const long long*)d_in[9];
  float* out = (float*)d_out;

  // workspace: W1 hi/lo + W2 hi/lo swizzled bf16 (4 x 2MB = 8MB)
  unsigned short* ws = (unsigned short*)d_ws;
  const size_t NW = (size_t)D_IN * D_HID;
  unsigned short* w1h = ws;
  unsigned short* w1l = ws + NW;
  unsigned short* w2h = ws + 2 * NW;
  unsigned short* w2l = ws + 3 * NW;

  int blocks = (int)((NW + 255) / 256);
  swizzle_w_kernel<<<blocks, 256, 0, stream>>>(W1, D_IN, D_HID, w1h, w1l);
  swizzle_w_kernel<<<blocks, 256, 0, stream>>>(W2, D_HID, D_IN, w2h, w2l);

  fused_mlp_moe_kernel<<<T_TOK / M_TILE, 256, 0, stream>>>(
      x, b1, b2, w_down, w_up, tkp, tki, w1h, w1l, w2h, w2l, out);
}
